// HeteroGAE_22574348107990
// MI455X (gfx1250) — matrix operand order repaired and checked by reading, not compiled, not run on hardware
//
#include <hip/hip_runtime.h>
#include <hip/hip_bf16.h>
#include <math.h>

#define N_NODES 100000
#define N_EDGES 1250000
#define DIM     64
#define N_BASES 16

typedef __attribute__((ext_vector_type(16))) __bf16 v16bf;
typedef __attribute__((ext_vector_type(8)))  float  v8f;

// ---------------------------------------------------------------------------
// Workspace layout (tiny: ~16.9 KB):
//   [0      , 8192 )  : B-fragments of W_msg  (4 ntile x 2 kstep x 32 lane x 16 bf16)
//   [8192   , 16384)  : B-fragments of W_self (same shape)
//   [16384  , 16640)  : b_msg  (64 f32)
//   [16640  , 16896)  : b_self (64 f32)
// B-fragment per-lane layout (16x16x32 bf16 WMMA, wave32):
//   lane = half*16 + n  (n = output col within 16-tile, half selects K range)
//   element j (0..15) of lane = W[k][n] with k = ks*32 + half*8 + j (j<8), ks*32 + 16 + half*8 + (j-8) (j>=8)
// ---------------------------------------------------------------------------

__device__ __forceinline__ void atomic_add_f32(float* p, float v) {
    // native global_atomic_add_f32 path (no CAS loop)
    unsafeAtomicAdd(p, v);
}

// Build a 16x32 bf16 A-fragment for one lane from a row-major f32 row.
// ISA 16-bit A layout: lanes 0-15 (half=0) hold K = {0..7, 16..23},
// lanes 16-31 (half=1) hold K = {8..15, 24..31}; element j<8 -> VGPR0..3,
// j>=8 -> VGPR4..7.
__device__ __forceinline__ v16bf make_afrag(const float* row, int ks, int half) {
    v16bf a;
    const float* p0 = row + ks * 32 + half * 8;        // K = ks*32 + half*8 + j
    const float* p1 = p0 + 16;                         // K = ks*32 + 16 + half*8 + j
#pragma unroll
    for (int j = 0; j < 8; ++j) a[j] = (__bf16)p0[j];
#pragma unroll
    for (int j = 0; j < 8; ++j) a[8 + j] = (__bf16)p1[j];
    return a;
}

// ---- Kernel 0: contract bases -> swizzled bf16 weight fragments + f32 biases
__global__ void __launch_bounds__(256)
k_prep(const float* __restrict__ wt_msg, const float* __restrict__ bias_msg_in,
       const float* __restrict__ wt_self, const float* __restrict__ bias_self_in,
       const float* __restrict__ lin_comb,
       __bf16* __restrict__ frag_msg, __bf16* __restrict__ frag_self,
       float* __restrict__ b_msg, float* __restrict__ b_self) {
    float c[N_BASES];
#pragma unroll
    for (int b = 0; b < N_BASES; ++b) c[b] = lin_comb[b];

    for (int idx = threadIdx.x; idx < DIM * DIM; idx += blockDim.x) {
        int k = idx >> 6;          // input dim (contracted, K of GEMM)
        int n = idx & 63;          // output dim (N of GEMM)
        const float* pm = wt_msg  + (size_t)idx * N_BASES;   // [i][o][b]
        const float* ps = wt_self + (size_t)idx * N_BASES;
        float vm = 0.f, vs = 0.f;
#pragma unroll
        for (int b = 0; b < N_BASES; ++b) { vm += pm[b] * c[b]; vs += ps[b] * c[b]; }
        int nt = n >> 4, nl = n & 15;
        int ks = k >> 5, kk = k & 31;
        int half = (kk >> 3) & 1, j = (kk & 7) + ((kk >> 4) << 3);
        int pos = (((nt * 2 + ks) * 32) + (half * 16 + nl)) * 16 + j;
        frag_msg[pos]  = (__bf16)vm;
        frag_self[pos] = (__bf16)vs;
    }
    if (threadIdx.x < DIM) {
        int o = threadIdx.x;
        float vm = 0.f, vs = 0.f;
#pragma unroll
        for (int b = 0; b < N_BASES; ++b) {
            vm += bias_msg_in[o * N_BASES + b] * c[b];
            vs += bias_self_in[o * N_BASES + b] * c[b];
        }
        b_msg[o] = vm;
        b_self[o] = vs;
    }
}

// ---- Kernel 1: out = x @ W_self + b_self   (plain stores; initializes d_out)
__global__ void __launch_bounds__(256)
k_self(const float* __restrict__ x, const __bf16* __restrict__ frag_self,
       const float* __restrict__ b_self, float* __restrict__ out) {
    int tile = blockIdx.x * 8 + (threadIdx.x >> 5);
    if (tile >= N_NODES / 16) return;
    int lane = threadIdx.x & 31;
    int half = lane >> 4;
    int nl   = lane & 15;                 // A row index AND C column index
    int base = tile * 16;

    const float* row = x + (size_t)(base + nl) * DIM;
    v16bf a0 = make_afrag(row, 0, half);
    v16bf a1 = make_afrag(row, 1, half);

#pragma unroll
    for (int nt = 0; nt < 4; ++nt) {
        float bias = b_self[nt * 16 + nl];
        v8f acc;
#pragma unroll
        for (int r = 0; r < 8; ++r) acc[r] = bias;
        v16bf b0 = *(const v16bf*)(frag_self + (((nt * 2 + 0) * 32) + lane) * 16);
        v16bf b1 = *(const v16bf*)(frag_self + (((nt * 2 + 1) * 32) + lane) * 16);
        acc = __builtin_amdgcn_wmma_f32_16x16x32_bf16(false, a0, false, b0,
                                                      (short)0, acc, false, false);
        acc = __builtin_amdgcn_wmma_f32_16x16x32_bf16(false, a1, false, b1,
                                                      (short)0, acc, false, false);
#pragma unroll
        for (int r = 0; r < 8; ++r)       // C/D: row M = r + 8*half, col N = nl
            out[(size_t)(base + half * 8 + r) * DIM + nt * 16 + nl] = acc[r];
    }
}

// ---- Kernel 2: per-edge messages: out[dst] += x[src] @ W_msg + b_msg
__global__ void __launch_bounds__(256)
k_msg(const float* __restrict__ x, const int* __restrict__ edge_index,
      const __bf16* __restrict__ frag_msg, const float* __restrict__ b_msg,
      float* __restrict__ out) {
    int tile = blockIdx.x * 8 + (threadIdx.x >> 5);
    if (tile >= N_EDGES / 16) return;
    int lane = threadIdx.x & 31;
    int half = lane >> 4;
    int nl   = lane & 15;
    int base = tile * 16;

    int src = edge_index[base + nl];                       // src row for A row nl
    const float* row = x + (size_t)src * DIM;              // gather (L2-resident)
    v16bf a0 = make_afrag(row, 0, half);
    v16bf a1 = make_afrag(row, 1, half);

    int dstRow[8];
#pragma unroll
    for (int r = 0; r < 8; ++r)
        dstRow[r] = edge_index[N_EDGES + base + half * 8 + r];

#pragma unroll
    for (int nt = 0; nt < 4; ++nt) {
        float bias = b_msg[nt * 16 + nl];
        v8f acc;
#pragma unroll
        for (int r = 0; r < 8; ++r) acc[r] = bias;
        v16bf b0 = *(const v16bf*)(frag_msg + (((nt * 2 + 0) * 32) + lane) * 16);
        v16bf b1 = *(const v16bf*)(frag_msg + (((nt * 2 + 1) * 32) + lane) * 16);
        acc = __builtin_amdgcn_wmma_f32_16x16x32_bf16(false, a0, false, b0,
                                                      (short)0, acc, false, false);
        acc = __builtin_amdgcn_wmma_f32_16x16x32_bf16(false, a1, false, b1,
                                                      (short)0, acc, false, false);
#pragma unroll
        for (int r = 0; r < 8; ++r)
            atomic_add_f32(out + (size_t)dstRow[r] * DIM + nt * 16 + nl, acc[r]);
    }
}

// ---- Kernel 3: in-place row-wise L2 normalize (one wave32 per 64-float row)
__global__ void __launch_bounds__(256)
k_norm(float* __restrict__ out) {
    int row = blockIdx.x * 8 + (threadIdx.x >> 5);
    if (row >= N_NODES) return;
    int lane = threadIdx.x & 31;
    float2* p = (float2*)(out + (size_t)row * DIM) + lane;
    float2 v = *p;
    float s = v.x * v.x + v.y * v.y;
#pragma unroll
    for (int off = 16; off >= 1; off >>= 1)
        s += __shfl_xor(s, off, 32);
    float inv = 1.0f / fmaxf(sqrtf(s), 1e-12f);
    v.x *= inv; v.y *= inv;
    *p = v;
}

extern "C" void kernel_launch(void* const* d_in, const int* in_sizes, int n_in,
                              void* d_out, int out_size, void* d_ws, size_t ws_size,
                              hipStream_t stream) {
    const float* x      = (const float*)d_in[0];
    const int*   ei     = (const int*)  d_in[1];   // [2, E] row-major int32
    const float* wmsg   = (const float*)d_in[2];
    const float* bmsg   = (const float*)d_in[3];
    const float* wself  = (const float*)d_in[4];
    const float* bself  = (const float*)d_in[5];
    const float* lc     = (const float*)d_in[6];
    float*       out    = (float*)d_out;

    char* ws = (char*)d_ws;
    __bf16* frag_msg  = (__bf16*)(ws);
    __bf16* frag_self = (__bf16*)(ws + 8192);
    float*  b_msg     = (float*) (ws + 16384);
    float*  b_self    = (float*) (ws + 16640);

    k_prep<<<1, 256, 0, stream>>>(wmsg, bmsg, wself, bself, lc,
                                  frag_msg, frag_self, b_msg, b_self);

    const int nodeTiles = N_NODES / 16;   // 6250 exactly
    k_self<<<(nodeTiles + 7) / 8, 256, 0, stream>>>(x, frag_self, b_self, out);

    const int edgeTiles = N_EDGES / 16;   // 78125 exactly
    k_msg<<<(edgeTiles + 7) / 8, 256, 0, stream>>>(x, ei, frag_msg, b_msg, out);

    k_norm<<<(N_NODES + 7) / 8, 256, 0, stream>>>(out);
}
